// CapsuleLayer_42829413876192
// MI455X (gfx1250) — matrix operand order repaired and checked by reading, not compiled, not run on hardware
//
#include <hip/hip_runtime.h>
#include <hip/hip_bf16.h>
#include <stdint.h>

// CapsuleLayer dynamic routing for MI455X (gfx1250, wave32).
// Stage 1: bf16 WMMA batched GEMM -> inputs_hat (bf16, [B,J,K,D'] with
//          interleaved d: stored d' = 2*ml + nt  <->  real d = 16*nt + ml)
// Stage 2: 3 routing iterations, fused into 3 streaming passes over hat.
//          All intermediates live in the permuted d' order; only the final
//          squash un-permutes when writing d_out.

typedef __bf16 bf16_t;
typedef __attribute__((ext_vector_type(16))) __bf16 v16bf;
typedef __attribute__((ext_vector_type(8)))  float  v8f;
typedef __attribute__((ext_vector_type(4)))  float  f4;

#define Bdim 32
#define Jdim 2048
#define Idim 64
#define Kdim 32
#define Ddim 32
#define WAVES 8
#define JPB 256   // j's per routing block

struct U8 { uint32_t u[8]; };

// Pack two f32 -> packed bf16 pair (a -> low half, b -> high half).
// Round-half-up (+0x8000) then v_perm_b32 grabs both high halves in one op.
static __device__ __forceinline__ uint32_t pkbf(float a, float b) {
    uint32_t ua = __builtin_bit_cast(uint32_t, a) + 0x8000u;
    uint32_t ub = __builtin_bit_cast(uint32_t, b) + 0x8000u;
    return __builtin_amdgcn_perm(ub, ua, 0x07060302u);
}
static __device__ __forceinline__ float bfu2f_lo(uint32_t u) {
    return __builtin_bit_cast(float, u << 16);
}
static __device__ __forceinline__ float bfu2f_hi(uint32_t u) {
    return __builtin_bit_cast(float, u & 0xFFFF0000u);
}
static __device__ __forceinline__ v16bf pack_frag(f4 f0, f4 f1, f4 f2, f4 f3) {
    U8 pk;
    pk.u[0] = pkbf(f0.x, f0.y);  pk.u[1] = pkbf(f0.z, f0.w);
    pk.u[2] = pkbf(f1.x, f1.y);  pk.u[3] = pkbf(f1.z, f1.w);
    pk.u[4] = pkbf(f2.x, f2.y);  pk.u[5] = pkbf(f2.z, f2.w);
    pk.u[6] = pkbf(f3.x, f3.y);  pk.u[7] = pkbf(f3.z, f3.w);
    return __builtin_bit_cast(v16bf, pk);
}

// ---------------------------------------------------------------------------
// Stage 1: inputs_hat[b,j,k,d'] = sum_i inputs[b,j,i] * W[k,j,d,i]
// One wave per (k,j): [32x64] x [32x64]^T -> [32x32] via 8 bf16 WMMAs.
// ---------------------------------------------------------------------------
__global__ __launch_bounds__(256)
void caps_hat_wmma(const float* __restrict__ inp,   // [B,J,I]
                   const float* __restrict__ W,     // [K,J,D,I]
                   bf16_t* __restrict__ hat)        // [B,J,K,D'] (d interleaved)
{
    const int lane = threadIdx.x & 31;
    const int wave = threadIdx.x >> 5;
    const int p    = blockIdx.x * WAVES + wave;     // 0 .. K*J-1
    const int k    = p >> 11;                       // / Jdim
    const int j    = p & (Jdim - 1);
    const int half = lane >> 4;
    const int ml   = lane & 15;

    // ---- A fragments: inputs rows (M=batch, K=i), ISA 16-bit A 16x32 layout:
    //  e in [0,8):  i = 8*half + e  ;  e in [8,16): i = 16 + 8*half + (e-8)
    v16bf afrag[2][2];
#pragma unroll
    for (int mt = 0; mt < 2; ++mt) {
        const int brow = ml + 16 * mt;
        const float* ap = inp + ((size_t)brow * Jdim + j) * Idim;
#pragma unroll
        for (int ks = 0; ks < 2; ++ks) {
            const float* p0 = ap + 32 * ks + 8 * half;
            const float* p1 = ap + 32 * ks + 16 + 8 * half;
            f4 f0 = *(const f4*)(p0);
            f4 f1 = *(const f4*)(p0 + 4);
            f4 f2 = *(const f4*)(p1);
            f4 f3 = *(const f4*)(p1 + 4);
            afrag[mt][ks] = pack_frag(f0, f1, f2, f3);
        }
    }

    // ---- B fragments: W rows (N=d, K=i), ISA 16-bit B 32x16 layout:
    //  e in [0,8):  i = 8*half + e  ;  e in [8,16): i = 16 + 8*half + (e-8)
    //  W is streamed exactly once -> non-temporal loads keep L2 for inp/hat.
    v16bf bfrag[2][2];
#pragma unroll
    for (int nt = 0; nt < 2; ++nt) {
        const int d = ml + 16 * nt;
        const float* wp = W + (((size_t)k * Jdim + j) * Ddim + d) * Idim;
#pragma unroll
        for (int ks = 0; ks < 2; ++ks) {
            const f4* p0 = (const f4*)(wp + 32 * ks + 8 * half);
            const f4* p1 = (const f4*)(wp + 32 * ks + 16 + 8 * half);
            f4 f0 = __builtin_nontemporal_load(p0);
            f4 f1 = __builtin_nontemporal_load(p0 + 1);
            f4 f2 = __builtin_nontemporal_load(p1);
            f4 f3 = __builtin_nontemporal_load(p1 + 1);
            bfrag[nt][ks] = pack_frag(f0, f1, f2, f3);
        }
    }

    v8f acc[2][2];
#pragma unroll
    for (int mt = 0; mt < 2; ++mt)
#pragma unroll
        for (int nt = 0; nt < 2; ++nt)
#pragma unroll
            for (int e = 0; e < 8; ++e) acc[mt][nt][e] = 0.0f;

#pragma unroll
    for (int ks = 0; ks < 2; ++ks)
#pragma unroll
        for (int mt = 0; mt < 2; ++mt)
#pragma unroll
            for (int nt = 0; nt < 2; ++nt)
                acc[mt][nt] = __builtin_amdgcn_wmma_f32_16x16x32_bf16(
                    false, afrag[mt][ks], false, bfrag[nt][ks],
                    (short)0, acc[mt][nt], false, false);

    // C/D layout: VGPR r -> M = 16*mt + 8*half + r, N = 16*nt + ml.
    // Lane holds (nt=0, nt=1) for fixed (M, ml): pack both into one b32 store
    // at interleaved position d' = 2*ml + nt.
    uint32_t* hb = (uint32_t*)(hat + (((size_t)(8 * half) * Jdim + j) * Kdim + k) * Ddim) + ml;
    const size_t rowStride = (size_t)Jdim * Kdim * Ddim / 2;  // u32 units per brow
#pragma unroll
    for (int mt = 0; mt < 2; ++mt)
#pragma unroll
        for (int r = 0; r < 8; ++r)
            hb[(size_t)(mt * 16 + r) * rowStride] = pkbf(acc[mt][0][r], acc[mt][1][r]);
}

// ---------------------------------------------------------------------------
// Stage 2: one routing pass. lane = k; per j: logits = U.hat, softmax over K
// (cross-lane within the wave32), accumulate c*hat into registers, then
// atomically fold into S[b,k,d']. U == nullptr -> uniform c (= 1, prescaled
// later by 1/K inside squash). d' order is consistent everywhere here.
// ---------------------------------------------------------------------------
__global__ __launch_bounds__(256)
void caps_route(const bf16_t* __restrict__ hat,   // [B,J,K,D']
                const float* __restrict__ U,      // [B,K,D'] or nullptr
                float* __restrict__ S)            // [B,K,D'] accumulated
{
    __shared__ float Ub[Kdim][Ddim + 1];          // pad to avoid bank conflicts
    const int b    = blockIdx.x;
    const int jblk = blockIdx.y;
    const int lane = threadIdx.x & 31;
    const int wave = threadIdx.x >> 5;
    const int k    = lane;

    if (U != nullptr) {
        for (int t = threadIdx.x; t < Kdim * Ddim; t += 256)
            Ub[t >> 5][t & 31] = U[(size_t)b * Kdim * Ddim + t];
        __syncthreads();
    }

    float sacc[Ddim];
#pragma unroll
    for (int d = 0; d < Ddim; ++d) sacc[d] = 0.0f;

    for (int jj = wave; jj < JPB; jj += WAVES) {
        const int j = jblk * JPB + jj;
        const bf16_t* hp = hat + (((size_t)b * Jdim + j) * Kdim + k) * Ddim;
        uint4 u0 = *(const uint4*)(hp);
        uint4 u1 = *(const uint4*)(hp + 8);
        uint4 u2 = *(const uint4*)(hp + 16);
        uint4 u3 = *(const uint4*)(hp + 24);
        uint32_t uw[16] = { u0.x,u0.y,u0.z,u0.w, u1.x,u1.y,u1.z,u1.w,
                            u2.x,u2.y,u2.z,u2.w, u3.x,u3.y,u3.z,u3.w };
        float hv[Ddim];
#pragma unroll
        for (int q = 0; q < 16; ++q) {
            hv[2 * q]     = bfu2f_lo(uw[q]);
            hv[2 * q + 1] = bfu2f_hi(uw[q]);
        }

        float c;
        if (U != nullptr) {
            float lg = 0.0f;
#pragma unroll
            for (int d = 0; d < Ddim; ++d) lg = fmaf(Ub[k][d], hv[d], lg);
            float m = lg;
#pragma unroll
            for (int off = 16; off > 0; off >>= 1)
                m = fmaxf(m, __shfl_xor(m, off, 32));
            float e = __expf(lg - m);
            float s = e;
#pragma unroll
            for (int off = 16; off > 0; off >>= 1)
                s += __shfl_xor(s, off, 32);
            c = e / s;
        } else {
            c = 1.0f;   // uniform routing; 1/K folded into squash prescale
        }
#pragma unroll
        for (int d = 0; d < Ddim; ++d) sacc[d] = fmaf(c, hv[d], sacc[d]);
    }

#pragma unroll
    for (int d = 0; d < Ddim; ++d)
        atomicAdd(&S[((size_t)b * Kdim + k) * Ddim + d], sacc[d]);
}

// ---------------------------------------------------------------------------
// Squash: V = squash(prescale * S).
// umode: 1 -> Usum = V ; 2 -> Usum += V ; 3 -> final write, un-permute d'.
// ---------------------------------------------------------------------------
__global__ __launch_bounds__(32)
void caps_squash(const float* __restrict__ S, float prescale,
                 float* __restrict__ V, float* __restrict__ Usum, int umode)
{
    const int bk = blockIdx.x;
    const int d  = threadIdx.x;                 // stored (permuted) index d'
    float s  = S[(size_t)bk * Ddim + d] * prescale;
    float n2 = s * s;
#pragma unroll
    for (int off = 16; off > 0; off >>= 1)
        n2 += __shfl_xor(n2, off, 32);
    float v = s * (n2 / ((1.0f + n2) * sqrtf(n2 + 1e-7f)));
    if (umode == 3) {
        const int dreal = ((d & 1) << 4) | (d >> 1);   // d' -> reference d
        V[(size_t)bk * Ddim + dreal] = v;
    } else {
        V[(size_t)bk * Ddim + d] = v;
        if (umode == 1)      Usum[(size_t)bk * Ddim + d]  = v;
        else if (umode == 2) Usum[(size_t)bk * Ddim + d] += v;
    }
}

__global__ void caps_zero(float* __restrict__ p, int n) {
    int i = blockIdx.x * blockDim.x + threadIdx.x;
    if (i < n) p[i] = 0.0f;
}

// ---------------------------------------------------------------------------
extern "C" void kernel_launch(void* const* d_in, const int* in_sizes, int n_in,
                              void* d_out, int out_size, void* d_ws, size_t ws_size,
                              hipStream_t stream) {
    (void)in_sizes; (void)n_in; (void)out_size;
    const float* inp = (const float*)d_in[0];   // [B,J,I]
    const float* W   = (const float*)d_in[1];   // [K,J,D,I]
    float* out = (float*)d_out;                 // [B,K,D]

    const size_t hatBytes = (size_t)Bdim * Jdim * Kdim * Ddim * sizeof(bf16_t); // 128 MB
    const int    BKD      = Bdim * Kdim * Ddim;                                  // 32768
    const size_t need     = hatBytes + 3 * (size_t)BKD * sizeof(float);
    if (ws_size < need) return;   // deterministic guard (ws_size fixed per run)

    bf16_t* hat  = (bf16_t*)d_ws;
    float*  S    = (float*)((char*)d_ws + hatBytes);
    float*  V    = S + BKD;
    float*  Usum = V + BKD;

    const dim3 rgrid(Bdim, Jdim / JPB);
    const int  zgrid = (BKD + 255) / 256;

    // Stage 1: inputs_hat via bf16 WMMA (memory-bound on W: ~512 MB stream)
    caps_hat_wmma<<<(Kdim * Jdim) / WAVES, 256, 0, stream>>>(inp, W, hat);

    // Iter 0: uniform c -> S0 = sum_j hat ; out0 = squash(S0 / K); Usum = out0
    caps_zero<<<zgrid, 256, 0, stream>>>(S, BKD);
    caps_route<<<rgrid, 256, 0, stream>>>(hat, nullptr, S);
    caps_squash<<<Bdim * Kdim, 32, 0, stream>>>(S, 1.0f / (float)Kdim, V, Usum, 1);

    // Iter 1: logits b1 = out0 . hat ; out1 = squash(sum_j softmax(b1)*hat)
    caps_zero<<<zgrid, 256, 0, stream>>>(S, BKD);
    caps_route<<<rgrid, 256, 0, stream>>>(hat, Usum, S);
    caps_squash<<<Bdim * Kdim, 32, 0, stream>>>(S, 1.0f, V, Usum, 2); // Usum = out0+out1

    // Iter 2: logits b2 = (out0+out1) . hat ; final output (un-permuted d)
    caps_zero<<<zgrid, 256, 0, stream>>>(S, BKD);
    caps_route<<<rgrid, 256, 0, stream>>>(hat, Usum, S);
    caps_squash<<<Bdim * Kdim, 32, 0, stream>>>(S, 1.0f, out, nullptr, 3);
}
